// FTTransformer_27178553049544
// MI455X (gfx1250) — matrix operand order repaired and checked by reading, not compiled, not run on hardware
//
#include <hip/hip_runtime.h>
#include <hip/hip_bf16.h>
#include <string.h>

// ---------------------------------------------------------------------------
// FT-Transformer forward on MI455X (gfx1250), bf16 WMMA, fp32 accumulation.
//
// Sizes: B=128, S=129 -> M = 16512 tokens (divisible by 32), EMB=512, H=8,
// hd=64, DEPTH=6, HID=682 (pad 2*HID 1364->1408, HID 682->704).
//
// GEMM: K-block count is a template constant so all fragment addresses are
// base + immediate offset; double-buffered fragment sets + __launch_bounds__
// give a spill-free software-pipelined inner loop.
// ---------------------------------------------------------------------------

#define BATCH   128
#define SEQ     129
#define MTOK    (BATCH * SEQ)      // 16512
#define EMBD    512
#define NHEAD   8
#define HDIM    64
#define DEPTH   6
#define NQKV    1536
#define NP0     1408               // padded 2*HID (1364 -> 22*64)
#define KP1     704                // padded HID (682 -> 22*32)
#define N0REAL  1364
#define HIDR    682
#define SQRT_EMB 22.62741699796952f

typedef __bf16 bf16;
typedef __attribute__((ext_vector_type(16))) __bf16 v16bf;
typedef __attribute__((ext_vector_type(8)))  __bf16 v8bf;
typedef __attribute__((ext_vector_type(8)))  float  v8f;

static __device__ __forceinline__ bf16 f2bf(float f) {
  unsigned u = __float_as_uint(f);
  u += 0x7FFFu + ((u >> 16) & 1u);              // round-to-nearest-even
  unsigned short h = (unsigned short)(u >> 16);
  union { unsigned short s; bf16 b; } cv; cv.s = h; return cv.b;
}

static __device__ __forceinline__ v8f vzero8() {
  v8f z;
#pragma unroll
  for (int i = 0; i < 8; ++i) z[i] = 0.0f;
  return z;
}

// A-fragment (16x32 bf16, row-major source):
// lane L: m = L&15; elems 0..15 -> K = (L>>4)*16 + e (contiguous).
static __device__ __forceinline__ v16bf load_afrag(const bf16* p0) {
  union { v16bf v; v8bf h[2]; } u;
  u.h[0] = *(const v8bf*)(p0);
  u.h[1] = *(const v8bf*)(p0 + 8);
  return u.v;
}

// B-fragment (32x16 bf16) packed contiguously: lane L slot s <->
// (k = (L>>4)*16 + s, n = L&15); per-lane 32 contiguous bytes.
static __device__ __forceinline__ v16bf load_bfrag(const bf16* p) {
  return *(const v16bf*)p;
}

static __device__ __forceinline__ v8f wmma_bf16(v16bf a, v16bf b, v8f c) {
  return __builtin_amdgcn_wmma_f32_16x16x32_bf16(false, a, false, b,
                                                 (short)0, c, false, false);
}

// ---------------------------------------------------------------------------
// Weight packer: fp32 [Ks x Ns] row-major -> bf16 fragment layout, padded to
// [Kp x Np], fragment order frag_id = nblock*(Kp/32) + kblock.
// ---------------------------------------------------------------------------
__global__ __launch_bounds__(256)
void pack_w_kernel(const float* __restrict__ W, bf16* __restrict__ dst,
                   int Ks, int Ns, int Kp, int Np) {
  int total = Kp * Np;
  int kbcnt = Kp >> 5;
  for (int i = blockIdx.x * blockDim.x + threadIdx.x; i < total;
       i += gridDim.x * blockDim.x) {
    int frag   = i >> 9;
    int within = i & 511;
    int lane = within >> 4, slot = within & 15;
    int nblock = frag / kbcnt, kblock = frag % kbcnt;
    int k = kblock * 32 + (lane >> 4) * 16 + slot;
    int n = nblock * 16 + (lane & 15);
    float v = (k < Ks && n < Ns) ? W[(size_t)k * Ns + n] : 0.0f;
    dst[i] = f2bf(v);
  }
}

__global__ __launch_bounds__(256)
void pad_b0_kernel(const float* __restrict__ b0, float* __restrict__ b0p) {
  int total = DEPTH * NP0;
  for (int i = blockIdx.x * blockDim.x + threadIdx.x; i < total;
       i += gridDim.x * blockDim.x) {
    int l = i / NP0, j = i % NP0;
    b0p[i] = (j < N0REAL) ? b0[l * N0REAL + j] : 0.0f;
  }
}

// ---------------------------------------------------------------------------
// Feature tokenizer: h[b,s,e] fp32.
// ---------------------------------------------------------------------------
__global__ __launch_bounds__(256)
void tokenizer_kernel(const float* __restrict__ x,
                      const float* __restrict__ tw,
                      const float* __restrict__ tb,
                      const float* __restrict__ ce,
                      float* __restrict__ H) {
  const int total = MTOK * EMBD;
  for (int i = blockIdx.x * blockDim.x + threadIdx.x; i < total;
       i += gridDim.x * blockDim.x) {
    int e  = i & (EMBD - 1);
    int bs = i >> 9;              // EMBD == 512
    int s  = bs % SEQ;
    int b  = bs / SEQ;
    float val;
    if (s == 0) {
      val = tw[e];                                        // CLS: tok_w[0]*1, bias 0
    } else if (s < 101) {
      val = tw[s * EMBD + e] * x[b * 128 + 28 + (s - 1)] + tb[(s - 1) * EMBD + e];
    } else {
      int j  = s - 101;
      int ci = (int)x[b * 128 + j] + j * 100;             // OFFSETS[j] = j*100
      val = ce[(size_t)ci * EMBD + e] + tb[(s - 1) * EMBD + e];
    }
    H[i] = val;
  }
}

// ---------------------------------------------------------------------------
// LayerNorm (or plain convert when apply==0) -> bf16. One wave per token row.
// ---------------------------------------------------------------------------
__global__ __launch_bounds__(256)
void ln_bf16_kernel(const float* __restrict__ H,
                    const float* __restrict__ g,
                    const float* __restrict__ bta,
                    bf16* __restrict__ out, int rows, int apply) {
  int wid  = (blockIdx.x * blockDim.x + threadIdx.x) >> 5;
  int lane = threadIdx.x & 31;
  if (wid >= rows) return;
  const float* xr = H + (size_t)wid * EMBD;
  float v[16], s = 0.0f, ss = 0.0f;
#pragma unroll
  for (int c = 0; c < 16; ++c) {
    float t = xr[c * 32 + lane];
    v[c] = t; s += t; ss += t * t;
  }
#pragma unroll
  for (int m = 16; m; m >>= 1) {
    s  += __shfl_xor(s, m, 32);
    ss += __shfl_xor(ss, m, 32);
  }
  float mean = s * (1.0f / EMBD);
  float var  = ss * (1.0f / EMBD) - mean * mean;
  float rstd = rsqrtf(var + 1e-5f);
  bf16* o = out + (size_t)wid * EMBD;
#pragma unroll
  for (int c = 0; c < 16; ++c) {
    int e = c * 32 + lane;
    float t = apply ? (v[c] - mean) * rstd * g[e] + bta[e] : v[c];
    o[e] = f2bf(t);
  }
}

// ---------------------------------------------------------------------------
// WMMA GEMM: C[M,Npad] = A[M,K](bf16,row-major) @ Bpacked + bias.
// One wave -> 32x64 tile (2 M-tiles x 4 N-tiles). KBLOCKS (= K/32, even) is a
// template constant so every fragment load is base + immediate offset.
// MODE 0: store fp32; 1: store bf16; 2: fp32 accumulate (residual).
// ---------------------------------------------------------------------------
template <int KBLOCKS, int MODE>
__global__ __launch_bounds__(256)
void gemm_bf16_kernel(const bf16* __restrict__ A,
                      const bf16* __restrict__ Bp,
                      const float* __restrict__ bias,
                      float* __restrict__ Cf, bf16* __restrict__ Cb,
                      int M2, int nstrips, int ldc) {
  constexpr int K = KBLOCKS * 32;
  int wid  = (blockIdx.x * blockDim.x + threadIdx.x) >> 5;
  int lane = threadIdx.x & 31;
  int mrow2  = wid / nstrips;
  int nstrip = wid % nstrips;
  if (mrow2 >= M2) return;
  const int m_base = mrow2 * 32;
  const int hi = lane >> 4, nn = lane & 15;

  v8f c0[4], c1[4];
#pragma unroll
  for (int t = 0; t < 4; ++t) { c0[t] = vzero8(); c1[t] = vzero8(); }

  // base pointers; everything else is an immediate offset
  const bf16* ar0 = A + (size_t)(m_base + nn) * K + hi * 16;
  const bf16* ar1 = ar0 + (size_t)16 * K;
  const bf16* bb  = Bp + ((size_t)nstrip * 4) * KBLOCKS * 512 + (size_t)lane * 16;

  // double-buffered fragment register sets
  v16bf a0x, a1x, bx0, bx1, bx2, bx3;          // set X
  v16bf a0y, a1y, by0, by1, by2, by3;          // set Y

  auto load_x = [&](int kb) {
    a0x = load_afrag(ar0 + kb * 32);
    a1x = load_afrag(ar1 + kb * 32);
    bx0 = load_bfrag(bb + (size_t)(0 * KBLOCKS + kb) * 512);
    bx1 = load_bfrag(bb + (size_t)(1 * KBLOCKS + kb) * 512);
    bx2 = load_bfrag(bb + (size_t)(2 * KBLOCKS + kb) * 512);
    bx3 = load_bfrag(bb + (size_t)(3 * KBLOCKS + kb) * 512);
  };
  auto load_y = [&](int kb) {
    a0y = load_afrag(ar0 + kb * 32);
    a1y = load_afrag(ar1 + kb * 32);
    by0 = load_bfrag(bb + (size_t)(0 * KBLOCKS + kb) * 512);
    by1 = load_bfrag(bb + (size_t)(1 * KBLOCKS + kb) * 512);
    by2 = load_bfrag(bb + (size_t)(2 * KBLOCKS + kb) * 512);
    by3 = load_bfrag(bb + (size_t)(3 * KBLOCKS + kb) * 512);
  };
  auto comp_x = [&]() {
    c0[0] = wmma_bf16(a0x, bx0, c0[0]); c1[0] = wmma_bf16(a1x, bx0, c1[0]);
    c0[1] = wmma_bf16(a0x, bx1, c0[1]); c1[1] = wmma_bf16(a1x, bx1, c1[1]);
    c0[2] = wmma_bf16(a0x, bx2, c0[2]); c1[2] = wmma_bf16(a1x, bx2, c1[2]);
    c0[3] = wmma_bf16(a0x, bx3, c0[3]); c1[3] = wmma_bf16(a1x, bx3, c1[3]);
  };
  auto comp_y = [&]() {
    c0[0] = wmma_bf16(a0y, by0, c0[0]); c1[0] = wmma_bf16(a1y, by0, c1[0]);
    c0[1] = wmma_bf16(a0y, by1, c0[1]); c1[1] = wmma_bf16(a1y, by1, c1[1]);
    c0[2] = wmma_bf16(a0y, by2, c0[2]); c1[2] = wmma_bf16(a1y, by2, c1[2]);
    c0[3] = wmma_bf16(a0y, by3, c0[3]); c1[3] = wmma_bf16(a1y, by3, c1[3]);
  };

  // pipelined loop: KBLOCKS is even and >= 2.
  load_x(0);
#pragma unroll
  for (int kb = 0; kb < KBLOCKS - 2; kb += 2) {
    load_y(kb + 1);        // issue next loads before consuming set X
    comp_x();
    load_x(kb + 2);
    comp_y();
  }
  load_y(KBLOCKS - 1);
  comp_x();
  comp_y();

#pragma unroll
  for (int nt = 0; nt < 4; ++nt) {
    int ncol = nstrip * 64 + nt * 16 + nn;
    float bv = bias ? bias[ncol] : 0.0f;
#pragma unroll
    for (int r = 0; r < 8; ++r) {
      int m0v = m_base + hi * 8 + r;     // C layout: lane n=L&15, m = r + hi*8
      int m1v = m0v + 16;
      float v0 = c0[nt][r] + bv, v1 = c1[nt][r] + bv;
      size_t i0 = (size_t)m0v * ldc + ncol;
      size_t i1 = (size_t)m1v * ldc + ncol;
      if (MODE == 0)      { Cf[i0] = v0;        Cf[i1] = v1; }
      else if (MODE == 1) { Cb[i0] = f2bf(v0);  Cb[i1] = f2bf(v1); }
      else                { Cf[i0] += v0;       Cf[i1] += v1; }
    }
  }
}

// ---------------------------------------------------------------------------
// Attention: one workgroup per (b,h); 9 waves, wave w owns query rows
// [16w,16w+16). q/k row-major and V^T staged in LDS; scores = q@k^T via WMMA,
// double softmax with half-wave shuffles, o = attn@V via WMMA.
// j padded to 160 (5 K-blocks of 32) with zero columns.
// ---------------------------------------------------------------------------
#define SP 160
#define ATTN_SMEM ((144*64 + 144*64 + 64*SP + 9*16*SP) * 2)

__global__ __launch_bounds__(288)
void attn_kernel(const bf16* __restrict__ qkv, bf16* __restrict__ obuf) {
  extern __shared__ char smem_raw[];
  bf16* q_s    = (bf16*)smem_raw;        // [144][64]
  bf16* k_s    = q_s + 144 * 64;         // [144][64]
  bf16* vt_s   = k_s + 144 * 64;         // [64][160]  (V transposed)
  bf16* attn_s = vt_s + 64 * SP;         // [9][16][160]

  const int b = blockIdx.x >> 3;
  const int h = blockIdx.x & 7;
  const int tid = threadIdx.x;
  const bf16 z = f2bf(0.0f);

  // stage q, k (rows >= 129 zeroed)
  for (int i = tid; i < 144 * 64; i += 288) {
    int s = i >> 6, d = i & 63;
    bf16 qv = z, kv = z;
    if (s < SEQ) {
      size_t base = ((size_t)(b * SEQ + s)) * NQKV + h * 192 + d;
      qv = qkv[base]; kv = qkv[base + 64];
    }
    q_s[i] = qv; k_s[i] = kv;
  }
  // stage V^T [d][s]
  for (int i = tid; i < 64 * SP; i += 288) {
    int d = i / SP, s = i % SP;
    vt_s[i] = (s < SEQ)
        ? qkv[((size_t)(b * SEQ + s)) * NQKV + h * 192 + 128 + d] : z;
  }
  __syncthreads();

  const int warp = tid >> 5, lane = tid & 31;
  const int m0 = warp * 16;
  const int hi = lane >> 4, nn = lane & 15;

  // ---- scores = (q @ k^T) * sqrt(EMB) ----
  v8f sc[9];
#pragma unroll
  for (int jt = 0; jt < 9; ++jt) {
    v8f acc = vzero8();
#pragma unroll
    for (int kb = 0; kb < 2; ++kb) {
      v16bf a  = load_afrag(q_s + (m0 + nn) * 64 + kb * 32 + hi * 16);
      v16bf bb = load_bfrag(k_s + (jt * 16 + nn) * 64 + kb * 32 + hi * 16);
      acc = wmma_bf16(a, bb, acc);
    }
#pragma unroll
    for (int r = 0; r < 8; ++r) acc[r] *= SQRT_EMB;
    sc[jt] = acc;
  }

  // ---- double softmax over j (129 valid) ----
#pragma unroll
  for (int r = 0; r < 8; ++r) {
    float mx = -1e30f;
#pragma unroll
    for (int jt = 0; jt < 9; ++jt)
      if (jt * 16 + nn < SEQ) mx = fmaxf(mx, sc[jt][r]);
    for (int m = 8; m; m >>= 1) mx = fmaxf(mx, __shfl_xor(mx, m, 16));
    float sm = 0.0f;
#pragma unroll
    for (int jt = 0; jt < 9; ++jt) {
      float e = (jt * 16 + nn < SEQ) ? __expf(sc[jt][r] - mx) : 0.0f;
      sc[jt][r] = e; sm += e;
    }
    for (int m = 8; m; m >>= 1) sm += __shfl_xor(sm, m, 16);
    float inv = 1.0f / sm;
    float mx2 = -1e30f;
#pragma unroll
    for (int jt = 0; jt < 9; ++jt) {
      float p = sc[jt][r] * inv; sc[jt][r] = p;
      if (jt * 16 + nn < SEQ) mx2 = fmaxf(mx2, p);
    }
    for (int m = 8; m; m >>= 1) mx2 = fmaxf(mx2, __shfl_xor(mx2, m, 16));
    float sm2 = 0.0f;
#pragma unroll
    for (int jt = 0; jt < 9; ++jt) {
      float e2 = (jt * 16 + nn < SEQ) ? __expf(sc[jt][r] - mx2) : 0.0f;
      sc[jt][r] = e2; sm2 += e2;
    }
    for (int m = 8; m; m >>= 1) sm2 += __shfl_xor(sm2, m, 16);
    float inv2 = 1.0f / sm2;
#pragma unroll
    for (int jt = 0; jt < 9; ++jt) sc[jt][r] *= inv2;
  }

  // ---- write attn tile (bf16) to this wave's LDS region ----
  bf16* aw = attn_s + warp * (16 * SP);
#pragma unroll
  for (int jt = 0; jt < 9; ++jt)
#pragma unroll
    for (int r = 0; r < 8; ++r) {
      int row = hi * 8 + r, col = jt * 16 + nn;
      aw[row * SP + col] = (col < SEQ) ? f2bf(sc[jt][r]) : z;
    }
#pragma unroll
  for (int r = 0; r < 8; ++r)
    aw[(hi * 8 + r) * SP + 144 + nn] = z;   // zero pad cols 144..159
  __syncthreads();

  // ---- o = attn @ V ----
  v8f oc[4];
#pragma unroll
  for (int t = 0; t < 4; ++t) oc[t] = vzero8();
#pragma unroll
  for (int kb = 0; kb < 5; ++kb) {
    v16bf a = load_afrag(aw + nn * SP + kb * 32 + hi * 16);
#pragma unroll
    for (int nt = 0; nt < 4; ++nt) {
      v16bf bb = load_bfrag(vt_s + (nt * 16 + nn) * SP + kb * 32 + hi * 16);
      oc[nt] = wmma_bf16(a, bb, oc[nt]);
    }
  }
#pragma unroll
  for (int nt = 0; nt < 4; ++nt)
#pragma unroll
    for (int r = 0; r < 8; ++r) {
      int m = m0 + hi * 8 + r;
      if (m < SEQ)
        obuf[((size_t)(b * SEQ + m)) * EMBD + h * HDIM + nt * 16 + nn] =
            f2bf(oc[nt][r]);
    }
}

// ---------------------------------------------------------------------------
// ReGLU: Hb[m,j<682] = G[m,j] * relu(G[m,682+j]); zero-pad cols 682..703.
// ---------------------------------------------------------------------------
__global__ __launch_bounds__(256)
void reglu_kernel(const float* __restrict__ Gf, bf16* __restrict__ Hb) {
  const int total = MTOK * KP1;
  for (int i = blockIdx.x * blockDim.x + threadIdx.x; i < total;
       i += gridDim.x * blockDim.x) {
    int m = i / KP1, j = i % KP1;
    float v = 0.0f;
    if (j < HIDR) {
      const float* gr = Gf + (size_t)m * NP0;
      v = gr[j] * fmaxf(gr[HIDR + j], 0.0f);
    }
    Hb[i] = f2bf(v);
  }
}

__global__ __launch_bounds__(256)
void extract_cls_kernel(const float* __restrict__ Hf, float* __restrict__ out) {
  int i = blockIdx.x * blockDim.x + threadIdx.x;
  if (i >= BATCH * EMBD) return;
  int b = i >> 9, e = i & 511;
  out[i] = Hf[(size_t)b * SEQ * EMBD + e];
}

// ---------------------------------------------------------------------------
// Host orchestration
// ---------------------------------------------------------------------------
extern "C" void kernel_launch(void* const* d_in, const int* in_sizes, int n_in,
                              void* d_out, int out_size, void* d_ws, size_t ws_size,
                              hipStream_t stream) {
  const float* x       = (const float*)d_in[0];
  const float* tok_w   = (const float*)d_in[1];
  const float* tok_b   = (const float*)d_in[2];
  const float* cat_emb = (const float*)d_in[3];
  const float* Wqkv    = (const float*)d_in[4];
  const float* bqkv    = (const float*)d_in[5];
  const float* Wout    = (const float*)d_in[6];
  const float* bout    = (const float*)d_in[7];
  const float* W0      = (const float*)d_in[8];
  const float* b0      = (const float*)d_in[9];
  const float* W1      = (const float*)d_in[10];
  const float* b1      = (const float*)d_in[11];
  const float* ln0_g   = (const float*)d_in[12];
  const float* ln0_b   = (const float*)d_in[13];
  const float* ln1_g   = (const float*)d_in[14];
  const float* ln1_b   = (const float*)d_in[15];

  char* base = (char*)d_ws;
  size_t off = 0;
  auto carve = [&](size_t bytes) -> char* {
    char* p = base + off;
    off = (off + bytes + 255) & ~(size_t)255;
    return p;
  };
  float* Hf    = (float*)carve((size_t)MTOK * EMBD * 4);
  bf16*  Xb    = (bf16*) carve((size_t)MTOK * EMBD * 2);
  bf16*  QKVb  = (bf16*) carve((size_t)MTOK * NQKV * 2);
  bf16*  Ob    = (bf16*) carve((size_t)MTOK * EMBD * 2);
  float* Gf    = (float*)carve((size_t)MTOK * NP0 * 4);
  bf16*  Hb    = (bf16*) carve((size_t)MTOK * KP1 * 2);
  bf16*  WqkvP = (bf16*) carve((size_t)DEPTH * EMBD * NQKV * 2);
  bf16*  WoutP = (bf16*) carve((size_t)DEPTH * EMBD * EMBD * 2);
  bf16*  W0P   = (bf16*) carve((size_t)DEPTH * EMBD * NP0 * 2);
  bf16*  W1P   = (bf16*) carve((size_t)DEPTH * KP1 * EMBD * 2);
  float* b0p   = (float*)carve((size_t)DEPTH * NP0 * 4);

  const int TPB = 256;
  auto blocks_for = [](long n, int tpb) { return (int)((n + tpb - 1) / tpb); };

  // --- pack weights (redone every call: deterministic, no caching) ---
  for (int l = 0; l < DEPTH; ++l) {
    pack_w_kernel<<<blocks_for((long)EMBD * NQKV, TPB), TPB, 0, stream>>>(
        Wqkv + (size_t)l * EMBD * NQKV, WqkvP + (size_t)l * EMBD * NQKV,
        EMBD, NQKV, EMBD, NQKV);
    pack_w_kernel<<<blocks_for((long)EMBD * EMBD, TPB), TPB, 0, stream>>>(
        Wout + (size_t)l * EMBD * EMBD, WoutP + (size_t)l * EMBD * EMBD,
        EMBD, EMBD, EMBD, EMBD);
    pack_w_kernel<<<blocks_for((long)EMBD * NP0, TPB), TPB, 0, stream>>>(
        W0 + (size_t)l * EMBD * N0REAL, W0P + (size_t)l * EMBD * NP0,
        EMBD, N0REAL, EMBD, NP0);
    pack_w_kernel<<<blocks_for((long)KP1 * EMBD, TPB), TPB, 0, stream>>>(
        W1 + (size_t)l * HIDR * EMBD, W1P + (size_t)l * KP1 * EMBD,
        HIDR, EMBD, KP1, EMBD);
  }
  pad_b0_kernel<<<blocks_for((long)DEPTH * NP0, TPB), TPB, 0, stream>>>(b0, b0p);

  // --- tokenizer ---
  tokenizer_kernel<<<blocks_for((long)MTOK * EMBD, TPB), TPB, 0, stream>>>(
      x, tok_w, tok_b, cat_emb, Hf);

  const int M2 = MTOK / 32;
  auto gemm_blocks = [&](int Npad) {
    long waves = (long)M2 * (Npad / 64);
    return (int)((waves * 32 + TPB - 1) / TPB);
  };

  (void)hipFuncSetAttribute((const void*)attn_kernel,
                            hipFuncAttributeMaxDynamicSharedMemorySize, ATTN_SMEM);

  const int ln_blocks = MTOK / 8;   // one wave per token row, 8 waves/block

  for (int l = 0; l < DEPTH; ++l) {
    // xin = (l==0) ? h : LN0(h)  -> bf16
    ln_bf16_kernel<<<ln_blocks, TPB, 0, stream>>>(
        Hf, ln0_g + l * EMBD, ln0_b + l * EMBD, Xb, MTOK, l > 0 ? 1 : 0);
    // qkv = xin @ Wqkv + bqkv  (bf16 out)
    gemm_bf16_kernel<16, 1><<<gemm_blocks(NQKV), TPB, 0, stream>>>(
        Xb, WqkvP + (size_t)l * EMBD * NQKV, bqkv + l * NQKV,
        nullptr, QKVb, M2, NQKV / 64, NQKV);
    // attention -> Ob (bf16)
    attn_kernel<<<BATCH * NHEAD, 288, ATTN_SMEM, stream>>>(QKVb, Ob);
    // h += o @ Wout + bout
    gemm_bf16_kernel<16, 2><<<gemm_blocks(EMBD), TPB, 0, stream>>>(
        Ob, WoutP + (size_t)l * EMBD * EMBD, bout + l * EMBD,
        Hf, nullptr, M2, EMBD / 64, EMBD);
    // r = LN1(h) -> bf16
    ln_bf16_kernel<<<ln_blocks, TPB, 0, stream>>>(
        Hf, ln1_g + l * EMBD, ln1_b + l * EMBD, Xb, MTOK, 1);
    // G = r @ W0 + b0  (fp32, padded N)
    gemm_bf16_kernel<16, 0><<<gemm_blocks(NP0), TPB, 0, stream>>>(
        Xb, W0P + (size_t)l * EMBD * NP0, b0p + l * NP0,
        Gf, nullptr, M2, NP0 / 64, NP0);
    // Hb = reglu(G)  (bf16, padded K)
    reglu_kernel<<<blocks_for((long)MTOK * KP1, TPB), TPB, 0, stream>>>(Gf, Hb);
    // h += Hb @ W1 + b1
    gemm_bf16_kernel<22, 2><<<gemm_blocks(EMBD), TPB, 0, stream>>>(
        Hb, W1P + (size_t)l * KP1 * EMBD, b1 + l * EMBD,
        Hf, nullptr, M2, EMBD / 64, EMBD);
  }

  extract_cls_kernel<<<blocks_for((long)BATCH * EMBD, TPB), TPB, 0, stream>>>(
      Hf, (float*)d_out);
}
